// SlotAttention_11287174054553
// MI455X (gfx1250) — matrix operand order repaired and checked by reading, not compiled, not run on hardware
//
#include <hip/hip_runtime.h>
#include <math.h>

// ---------------------------------------------------------------------------
// Slot Attention for MI455X (gfx1250, wave32, WMMA).
// Memory-bound: keys/values materialized once as f16 (128MB, L2-resident),
// 3 fused attention passes. All matmuls via v_wmma_f32_16x16x32_f16.
// kv global layout is chunk-blocked (128 rows): [keys n x d | values d x n]
// so every WMMA fragment load/store is a 16-byte vector op.
// ---------------------------------------------------------------------------

typedef __attribute__((ext_vector_type(16))) _Float16 v16h;
typedef __attribute__((ext_vector_type(8)))  _Float16 v8h;
typedef __attribute__((ext_vector_type(8)))  float    v8f;
typedef __attribute__((ext_vector_type(4)))  float    v4f;

union F16x16 { v16h v; v8h h[2]; };

#define WMMA_F16(A, B, C) \
  __builtin_amdgcn_wmma_f32_16x16x32_f16(false, (A), false, (B), (short)0, (C), false, false)

namespace cfg {
constexpr int B = 32, N = 16384, S = 8, D = 64, H = 128;
constexpr int ROWS = 128;              // rows per chunk / block
constexpr int CHUNK = ROWS * 2 * D;    // 16384 halfs per chunk (16KB keys + 16KB valuesT)
}

// ---------------------------------------------------------------------------
// init: slots = mu + exp(log_sigma)*noise ;
// wkvSwz = [Wk.T|Wv.T] f16 pre-swizzled into WMMA B-fragment order:
//   frag f = ct*2+kc ; lane L ; elem j  <=  W[kc*32 + (L>>4)*16 + j][ct*16 + (L&15)]
// ---------------------------------------------------------------------------
__global__ __launch_bounds__(256) void init_kernel(
    const float* __restrict__ noise, const float* __restrict__ mu,
    const float* __restrict__ lsg, const float* __restrict__ Wk,
    const float* __restrict__ Wv, float* __restrict__ slots,
    _Float16* __restrict__ wkvSwz) {
  int idx = blockIdx.x * 256 + threadIdx.x;
  if (idx < cfg::B * cfg::S * cfg::D) {
    int rem = idx % (cfg::S * cfg::D);
    slots[idx] = mu[rem] + __expf(lsg[rem]) * noise[idx];
  }
  if (idx < 16 * 32 * 16) {  // 8192 swizzled weight elements
    int f = idx >> 9, L = (idx >> 4) & 31, j = idx & 15;
    int ct = f >> 1, kc = f & 1;
    int k = kc * 32 + (L >> 4) * 16 + j;
    int n = ct * 16 + (L & 15);
    float w = (n < 64) ? Wk[n * 64 + k] : Wv[(n - 64) * 64 + k];
    wkvSwz[idx] = (_Float16)w;
  }
}

// ---------------------------------------------------------------------------
// proj_kv: LN(x) @ [Wk.T|Wv.T] -> kv chunks. 128 rows/block, 8 waves,
// each wave owns one 16-row A tile. 16 WMMA per wave. Keys staged via LDS
// (coalesced b128 flush); values written transposed as one b128 per lane.
// ---------------------------------------------------------------------------
__global__ __launch_bounds__(256) void proj_kv_kernel(
    const float* __restrict__ x, const float* __restrict__ lng,
    const float* __restrict__ lnb, const _Float16* __restrict__ wkvSwz,
    _Float16* __restrict__ kv) {
  __shared__ _Float16 nT[128 * 72];   // normalized x f16; later reused for keys out
  int t = threadIdx.x;
  long rowbase = (long)blockIdx.x * cfg::ROWS;
  _Float16* kglob = kv + (size_t)blockIdx.x * cfg::CHUNK;        // keys [n][d]
  _Float16* vglob = kglob + cfg::ROWS * cfg::D;                  // values [d][n]

  // Each thread loads half a row of x (coalesced), LN via pair shuffle.
  int r = t >> 1, part = (t & 1) * 32;
  const float* xp = x + (rowbase + r) * 64 + part;
  float v[32], s = 0.f, s2 = 0.f;
#pragma unroll
  for (int i = 0; i < 8; i++) {
    v4f q = ((const v4f*)xp)[i];
#pragma unroll
    for (int u = 0; u < 4; u++) { float f = q[u]; v[i * 4 + u] = f; s += f; s2 += f * f; }
  }
  __builtin_prefetch(x + (rowbase + cfg::ROWS) * 64 + t * 32, 0, 1);
  s  += __shfl_xor(s, 1, 32);
  s2 += __shfl_xor(s2, 1, 32);
  float mean = s * (1.f / 64.f);
  float var  = s2 * (1.f / 64.f) - mean * mean;
  float rs   = rsqrtf(var + 1e-5f);
#pragma unroll
  for (int i8 = 0; i8 < 4; i8++) {   // vectorized f16 stores to LDS
    v8h pk;
#pragma unroll
    for (int u = 0; u < 8; u++) {
      int c = part + i8 * 8 + u;
      pk[u] = (_Float16)((v[i8 * 8 + u] - mean) * rs * lng[c] + lnb[c]);
    }
    *(v8h*)(&nT[r * 72 + part + i8 * 8]) = pk;
  }
  __syncthreads();

  int lane = t & 31, wv = t >> 5;
  int m = lane & 15, half = lane >> 4, koff = half * 16, moff = half * 8;

  // A fragments: contiguous 8-half runs -> ds_load_b128 pairs
  F16x16 a0, a1;
  {
    const _Float16* nrow = &nT[(wv * 16 + m) * 72];
    a0.h[0] = *(const v8h*)(nrow + koff);
    a0.h[1] = *(const v8h*)(nrow + 8 + koff);
    a1.h[0] = *(const v8h*)(nrow + 32 + koff);
    a1.h[1] = *(const v8h*)(nrow + 40 + koff);
  }
  __syncthreads();   // nT now reusable as keys staging tile

  const v8h* wf = (const v8h*)wkvSwz;
  for (int ct = 0; ct < 8; ct++) {  // 8 output column tiles of 16
    F16x16 b0, b1;
    int f0 = ct * 2, f1 = ct * 2 + 1;
    b0.h[0] = wf[(f0 * 32 + lane) * 2];
    b0.h[1] = wf[(f0 * 32 + lane) * 2 + 1];
    b1.h[0] = wf[(f1 * 32 + lane) * 2];
    b1.h[1] = wf[(f1 * 32 + lane) * 2 + 1];
    v8f c = {};
    c = WMMA_F16(a0.v, b0.v, c);
    c = WMMA_F16(a1.v, b1.v, c);
    if (ct < 4) {        // keys: stage in LDS (C rows are strided)
#pragma unroll
      for (int j = 0; j < 8; j++)
        nT[(wv * 16 + moff + j) * 72 + ct * 16 + m] = (_Float16)c[j];
    } else {             // values: transposed layout -> rows contiguous
      v8h pk;
#pragma unroll
      for (int j = 0; j < 8; j++) pk[j] = (_Float16)c[j];
      int d = (ct - 4) * 16 + m;
      *(v8h*)(&vglob[d * cfg::ROWS + wv * 16 + moff]) = pk;
    }
  }
  __syncthreads();
  // flush keys tile: fully coalesced 16B stores
#pragma unroll
  for (int ch = 0; ch < 8; ch++) {
    int gl = ch * 2048 + t * 8;
    ((v8h*)kglob)[ch * 256 + t] = *(const v8h*)(&nT[(gl >> 6) * 72 + (gl & 63)]);
  }
}

// ---------------------------------------------------------------------------
// slots_to_q: q = LN(slots) @ Wq.T * D^-0.5, written pre-swizzled as WMMA
// B-fragments qSwz[b][kc(2)][lane(32)][16], padded slot rows (s+8) zeroed.
// ---------------------------------------------------------------------------
__global__ __launch_bounds__(256) void slots_to_q_kernel(
    const float* __restrict__ slots, const float* __restrict__ g,
    const float* __restrict__ bt, const float* __restrict__ Wq,
    _Float16* __restrict__ qSwz) {
  int t = threadIdx.x;  // row = b*8 + s
  int b = t >> 3, sI = t & 7;
  const float* sp = slots + t * 64;
  float xn[64], s = 0.f, s2 = 0.f;
#pragma unroll
  for (int k = 0; k < 64; k++) { float f = sp[k]; xn[k] = f; s += f; s2 += f * f; }
  float mean = s / 64.f, var = s2 / 64.f - mean * mean, rs = rsqrtf(var + 1e-5f);
#pragma unroll
  for (int k = 0; k < 64; k++) xn[k] = (xn[k] - mean) * rs * g[k] + bt[k];
  const float scale = 0.125f;  // 64^-0.5
  float q[64];
  for (int d = 0; d < 64; d++) {
    float acc = 0.f;
    const float* wr = Wq + d * 64;
#pragma unroll
    for (int k = 0; k < 64; k++) acc += wr[k] * xn[k];
    q[d] = acc * scale;
  }
  _Float16* qb = qSwz + (size_t)b * 1024;
#pragma unroll
  for (int kc = 0; kc < 2; kc++)
#pragma unroll
    for (int hf = 0; hf < 2; hf++) {
      _Float16* dst = qb + (kc * 32 + hf * 16 + sI) * 16;      // lane = hf*16 + sI
      _Float16* dz  = qb + (kc * 32 + hf * 16 + sI + 8) * 16;  // padded lane
      for (int j = 0; j < 16; j++) {
        dst[j] = (_Float16)q[kc * 32 + hf * 16 + j];
        dz[j]  = (_Float16)0.f;
      }
    }
}

// ---------------------------------------------------------------------------
// attn_updates: logits WMMA -> masked softmax over S (8-lane butterfly in the
// C-fragment layout) -> attn^T @ values WMMA -> LDS reduce -> f32 atomics.
// One block = one (batch, 128-row chunk). 32 WMMA per block; every fragment
// load is an aligned b128.
// ---------------------------------------------------------------------------
__global__ __launch_bounds__(256) void attn_updates_kernel(
    const _Float16* __restrict__ kv, const _Float16* __restrict__ qSwz,
    float* __restrict__ updAcc, float* __restrict__ denomAcc) {
  __shared__ _Float16 kT[128 * 72];     // keys [n][d]
  __shared__ _Float16 vT[64 * 136];     // values [d][n]
  __shared__ _Float16 attnT[16 * 136];  // [s][n] (s rows 8..15 zero)
  __shared__ float    updT[2 * 16 * 64];
  int t = threadIdx.x;
  int b = blockIdx.x >> 7;
  const _Float16* cbase = kv + (size_t)blockIdx.x * cfg::CHUNK;

  {  // keys + valuesT tiles: fully coalesced 16B loads
    const v8h* srck = (const v8h*)cbase;
    const v8h* srcv = (const v8h*)(cbase + cfg::ROWS * cfg::D);
#pragma unroll
    for (int ch = 0; ch < 8; ch++) {
      int gl = ch * 2048 + t * 8;
      *(v8h*)(&kT[(gl >> 6) * 72 + (gl & 63)]) = srck[ch * 256 + t];
    }
#pragma unroll
    for (int ch = 0; ch < 8; ch++) {
      int gl = ch * 2048 + t * 8;
      *(v8h*)(&vT[(gl >> 7) * 136 + (gl & 127)]) = srcv[ch * 256 + t];
    }
  }
  __builtin_prefetch(kv + (size_t)(blockIdx.x + 1) * cfg::CHUNK + t * 128, 0, 1);
  __syncthreads();

  int lane = t & 31, wv = t >> 5;
  int ln15 = lane & 15, half = lane >> 4, koff = half * 16, moff = half * 8;

  // ---- logits[n (16 rows of this wave)][s (16, padded)] ----
  v8f lg = {};
  {
    const v8h* qf = (const v8h*)(qSwz + (size_t)b * 1024);
    F16x16 a0, a1, b0, b1;
    const _Float16* krow = &kT[(wv * 16 + ln15) * 72];
    a0.h[0] = *(const v8h*)(krow + koff);
    a0.h[1] = *(const v8h*)(krow + 8 + koff);
    a1.h[0] = *(const v8h*)(krow + 32 + koff);
    a1.h[1] = *(const v8h*)(krow + 40 + koff);
    b0.h[0] = qf[lane * 2];
    b0.h[1] = qf[lane * 2 + 1];
    b1.h[0] = qf[(32 + lane) * 2];
    b1.h[1] = qf[(32 + lane) * 2 + 1];
    lg = WMMA_F16(a0.v, b0.v, lg);
    lg = WMMA_F16(a1.v, b1.v, lg);
  }

  // ---- softmax over s=0..7 (aligned 8-lane groups), + EPS; padded s -> 0 ----
  float p[8];
  {
    float mx[8], sm[8];
#pragma unroll
    for (int j = 0; j < 8; j++) mx[j] = lg[j];
#pragma unroll
    for (int msk = 1; msk < 8; msk <<= 1)
#pragma unroll
      for (int j = 0; j < 8; j++) mx[j] = fmaxf(mx[j], __shfl_xor(mx[j], msk, 32));
#pragma unroll
    for (int j = 0; j < 8; j++) { p[j] = __expf(lg[j] - mx[j]); sm[j] = p[j]; }
#pragma unroll
    for (int msk = 1; msk < 8; msk <<= 1)
#pragma unroll
      for (int j = 0; j < 8; j++) sm[j] += __shfl_xor(sm[j], msk, 32);
#pragma unroll
    for (int j = 0; j < 8; j++) {
      p[j] = p[j] / sm[j] + 1e-8f;
      if (ln15 >= 8) p[j] = 0.f;  // mask padded slot columns
    }
  }

  // ---- denominator: sum_n attn[n][s] ----
  {
    float ds = 0.f;
#pragma unroll
    for (int j = 0; j < 8; j++) ds += p[j];
    ds += __shfl_xor(ds, 16, 32);  // combine both row halves of the tile
    if (half == 0 && ln15 < 8) atomicAdd(&denomAcc[b * 16 + ln15], ds);
  }

  // ---- stash attn transposed: 8 consecutive rows -> one b128 store ----
  {
    v8h pr;
#pragma unroll
    for (int j = 0; j < 8; j++) pr[j] = (_Float16)p[j];
    *(v8h*)(&attnT[ln15 * 136 + wv * 16 + moff]) = pr;
  }
  __syncthreads();

  // ---- updates[s][d] += attn^T @ values over this 128-row chunk ----
  {
    int dt = wv & 3, kh = wv >> 2;  // 4 d-tiles x 2 k-halves across 8 waves
    v8f acc = {};
#pragma unroll
    for (int kc = kh * 2; kc < kh * 2 + 2; kc++) {
      F16x16 aA, bV;
      const _Float16* arow = &attnT[ln15 * 136 + kc * 32];
      aA.h[0] = *(const v8h*)(arow + koff);
      aA.h[1] = *(const v8h*)(arow + 8 + koff);
      const _Float16* vrow = &vT[(dt * 16 + ln15) * 136 + kc * 32 + half * 16];
      bV.h[0] = *(const v8h*)(vrow);
      bV.h[1] = *(const v8h*)(vrow + 8);
      acc = WMMA_F16(aA.v, bV.v, acc);
    }
#pragma unroll
    for (int j = 0; j < 8; j++)  // disjoint per kh copy: plain stores
      updT[kh * 1024 + (moff + j) * 64 + dt * 16 + ln15] = acc[j];
  }
  __syncthreads();
#pragma unroll
  for (int i = 0; i < 4; i++) {
    int idx = i * 256 + t;
    atomicAdd(&updAcc[b * 1024 + idx], updT[idx] + updT[1024 + idx]);
  }
}

// ---------------------------------------------------------------------------
// slot_update: normalize updates, GRU cell, residual LN-MLP. 256 rows, scalar f32.
// ---------------------------------------------------------------------------
__global__ __launch_bounds__(256) void slot_update_kernel(
    const float* __restrict__ updAcc, const float* __restrict__ denomAcc,
    float* __restrict__ slots, const float* __restrict__ W_ih,
    const float* __restrict__ W_hh, const float* __restrict__ b_ih,
    const float* __restrict__ b_hh, const float* __restrict__ gM,
    const float* __restrict__ bM, const float* __restrict__ W1,
    const float* __restrict__ b1, const float* __restrict__ W2,
    const float* __restrict__ b2, float* __restrict__ outp, int write_out) {
  int t = threadIdx.x;
  int b = t >> 3, sI = t & 7;
  float upd[64], prev[64];
  float den = denomAcc[b * 16 + sI];
  const float* ua = updAcc + (b * 16 + sI) * 64;
  float* srow = slots + t * 64;
#pragma unroll
  for (int d = 0; d < 64; d++) { upd[d] = ua[d] / den; prev[d] = srow[d]; }

  float r[64], z[64], h[64];
  for (int d = 0; d < 64; d++) {
    float gi = b_ih[d], gh = b_hh[d];
    const float *wi = W_ih + d * 64, *wh = W_hh + d * 64;
    for (int k = 0; k < 64; k++) { gi += wi[k] * upd[k]; gh += wh[k] * prev[k]; }
    r[d] = 1.f / (1.f + __expf(-(gi + gh)));
  }
  for (int d = 0; d < 64; d++) {
    float gi = b_ih[64 + d], gh = b_hh[64 + d];
    const float *wi = W_ih + (64 + d) * 64, *wh = W_hh + (64 + d) * 64;
    for (int k = 0; k < 64; k++) { gi += wi[k] * upd[k]; gh += wh[k] * prev[k]; }
    z[d] = 1.f / (1.f + __expf(-(gi + gh)));
  }
  for (int d = 0; d < 64; d++) {
    float gi = b_ih[128 + d], gh = b_hh[128 + d];
    const float *wi = W_ih + (128 + d) * 64, *wh = W_hh + (128 + d) * 64;
    for (int k = 0; k < 64; k++) { gi += wi[k] * upd[k]; gh += wh[k] * prev[k]; }
    float nn = tanhf(gi + r[d] * gh);
    h[d] = (1.f - z[d]) * nn + z[d] * prev[d];
  }

  float s = 0.f, s2 = 0.f;
  for (int d = 0; d < 64; d++) { s += h[d]; s2 += h[d] * h[d]; }
  float mean = s / 64.f, var = s2 / 64.f - mean * mean, rs = rsqrtf(var + 1e-5f);
  float t2[64], outv[64];
  for (int d = 0; d < 64; d++) t2[d] = (h[d] - mean) * rs * gM[d] + bM[d];
  for (int d = 0; d < 64; d++) outv[d] = h[d] + b2[d];
  for (int j = 0; j < 128; j++) {
    float acc = b1[j];
    const float* w1 = W1 + j * 64;
    for (int k = 0; k < 64; k++) acc += w1[k] * t2[k];
    acc = fmaxf(acc, 0.f);
    for (int d = 0; d < 64; d++) outv[d] += W2[d * 128 + j] * acc;
  }
  for (int d = 0; d < 64; d++) {
    srow[d] = outv[d];
    if (write_out) outp[t * 64 + d] = outv[d];
  }
}

// ---------------------------------------------------------------------------
extern "C" void kernel_launch(void* const* d_in, const int* in_sizes, int n_in,
                              void* d_out, int out_size, void* d_ws, size_t ws_size,
                              hipStream_t stream) {
  (void)in_sizes; (void)n_in; (void)out_size; (void)ws_size;
  const float* x       = (const float*)d_in[0];
  const float* noise   = (const float*)d_in[1];
  const float* mu      = (const float*)d_in[2];
  const float* lsg     = (const float*)d_in[3];
  const float* ln_in_g = (const float*)d_in[4];
  const float* ln_in_b = (const float*)d_in[5];
  const float* ln_sl_g = (const float*)d_in[6];
  const float* ln_sl_b = (const float*)d_in[7];
  const float* ln_ml_g = (const float*)d_in[8];
  const float* ln_ml_b = (const float*)d_in[9];
  const float* Wq      = (const float*)d_in[10];
  const float* Wk      = (const float*)d_in[11];
  const float* Wv      = (const float*)d_in[12];
  const float* W_ih    = (const float*)d_in[13];
  const float* W_hh    = (const float*)d_in[14];
  const float* b_ih    = (const float*)d_in[15];
  const float* b_hh    = (const float*)d_in[16];
  const float* W1      = (const float*)d_in[17];
  const float* b1      = (const float*)d_in[18];
  const float* W2      = (const float*)d_in[19];
  const float* b2      = (const float*)d_in[20];

  char* w = (char*)d_ws;
  size_t off = 0;
  _Float16* kvbuf  = (_Float16*)(w + off); off += (size_t)cfg::B * cfg::N * 2 * cfg::D * 2;  // 128 MB
  _Float16* wkvSwz = (_Float16*)(w + off); off += 16 * 32 * 16 * 2;
  _Float16* qSwz   = (_Float16*)(w + off); off += cfg::B * 1024 * 2;
  float* slots     = (float*)(w + off);    off += cfg::B * 8 * 64 * 4;
  float* updAcc    = (float*)(w + off);    off += cfg::B * 16 * 64 * 4;
  float* denomAcc  = (float*)(w + off);    off += cfg::B * 16 * 4;

  init_kernel<<<64, 256, 0, stream>>>(noise, mu, lsg, Wk, Wv, slots, wkvSwz);
  proj_kv_kernel<<<cfg::B * cfg::N / cfg::ROWS, 256, 0, stream>>>(x, ln_in_g, ln_in_b, wkvSwz, kvbuf);

  for (int it = 0; it < 3; it++) {
    (void)hipMemsetAsync(updAcc, 0, (cfg::B * 16 * 64 + cfg::B * 16) * sizeof(float), stream);
    slots_to_q_kernel<<<1, 256, 0, stream>>>(slots, ln_sl_g, ln_sl_b, Wq, qSwz);
    attn_updates_kernel<<<cfg::B * cfg::N / cfg::ROWS, 256, 0, stream>>>(kvbuf, qSwz, updAcc, denomAcc);
    slot_update_kernel<<<1, 256, 0, stream>>>(updAcc, denomAcc, slots, W_ih, W_hh, b_ih, b_hh,
                                              ln_ml_g, ln_ml_b, W1, b1, W2, b2,
                                              (float*)d_out, it == 2 ? 1 : 0);
  }
}
